// LongformerAttention_9680856285571
// MI455X (gfx1250) — matrix operand order repaired and checked by reading, not compiled, not run on hardware
//
#include <hip/hip_runtime.h>

typedef __attribute__((ext_vector_type(16))) __bf16 v16bf;
typedef __attribute__((ext_vector_type(8)))  float  v8f;
typedef __attribute__((ext_vector_type(4)))  unsigned int u32x4;
typedef __attribute__((ext_vector_type(4)))  int i32x4;
typedef __attribute__((ext_vector_type(8)))  int i32x8;

union BFrag { u32x4 u4[2]; v16bf v; };

__device__ __forceinline__ unsigned short f2bf(float f) {
    unsigned u = __float_as_uint(f);
    u += 0x7FFFu + ((u >> 16) & 1u);   // round-to-nearest-even
    return (unsigned short)(u >> 16);
}

#define S_LEN   4096
#define NH      12
#define DHEAD   64
#define DMODEL  768
#define CHUNK   256
#define WINHALF 256
#define NKEYS   768     /* 3*CHUNK window */

// ---------------------------------------------------------------------------
// TDM: async 2-D tile load (global -> LDS), data_size = 2 bytes (bf16).
// D# packing per CDNA5 ISA ch.8 (group0: count/lds/global/type, group1: dims).
// ---------------------------------------------------------------------------
__device__ __forceinline__ void tdm_load_2d_b16(unsigned lds_off, const void* gaddr,
                                                unsigned tile0, unsigned tile1,
                                                unsigned long long stride0_elems,
                                                unsigned dim0, unsigned dim1)
{
#if __has_builtin(__builtin_amdgcn_tensor_load_to_lds)
    unsigned long long ga = (unsigned long long)gaddr;
    u32x4 g0;
    g0[0] = 1u;                                              // count=1, user mode
    g0[1] = lds_off;                                         // LDS byte address
    g0[2] = (unsigned)(ga & 0xFFFFFFFFu);                    // global_addr[31:0]
    g0[3] = (unsigned)((ga >> 32) & 0x01FFFFFFu) | (2u << 30); // addr[56:32] | type=2
    i32x8 g1;
    g1[0] = (int)(1u << 16);                                 // data_size=1 (2 bytes)
    g1[1] = (int)((dim0 & 0xFFFFu) << 16);                   // tensor_dim0[15:0]
    g1[2] = (int)((dim0 >> 16) | ((dim1 & 0xFFFFu) << 16));  // dim0[31:16] | dim1[15:0]
    g1[3] = (int)((dim1 >> 16) | (tile0 << 16));             // dim1[31:16] | tile_dim0
    g1[4] = (int)(tile1 & 0xFFFFu);                          // tile_dim1 (tile_dim2=0)
    g1[5] = (int)(stride0_elems & 0xFFFFFFFFull);            // tensor_dim0_stride[31:0]
    g1[6] = (int)((unsigned)(stride0_elems >> 32) & 0xFFFFu);// stride[47:32]
    g1[7] = 0;
    i32x4 z4 = {0, 0, 0, 0};
#if defined(__clang_major__) && __clang_major__ >= 23
    i32x8 z8 = {0, 0, 0, 0, 0, 0, 0, 0};
    __builtin_amdgcn_tensor_load_to_lds(g0, g1, z4, z4, z8, 0);
#else
    __builtin_amdgcn_tensor_load_to_lds(g0, g1, z4, z4, 0);
#endif
#else
    (void)lds_off; (void)gaddr; (void)tile0; (void)tile1;
    (void)stride0_elems; (void)dim0; (void)dim1;
#endif
}

__device__ __forceinline__ void tdm_wait0() {
#if __has_builtin(__builtin_amdgcn_s_wait_tensorcnt)
    __builtin_amdgcn_s_wait_tensorcnt(0);
#endif
}

__device__ __forceinline__ unsigned lds_byte_offset(const void* p) {
    // generic pointers to LDS: low 32 bits are the LDS byte offset
    return (unsigned)(unsigned long long)p;
}

// ---------------------------------------------------------------------------
// Kernel 1: projection  X(8192x768) @ W(768x768) + bias -> bf16 (q scaled)
// grid (128 row-tiles of 64, 12 col-tiles of 64 == heads), block 128 (4 waves)
// transposed==0: out[((b*NH+h)*S + s)*DH + d]
// transposed==1: out[((b*NH+h)*DH + d)*S + s]   (for V, so PV B-frags are contiguous)
// ---------------------------------------------------------------------------
__global__ __launch_bounds__(128)
void qkv_proj_kernel(const float* __restrict__ X, const float* __restrict__ W,
                     const float* __restrict__ bias, unsigned short* __restrict__ out,
                     float scale, int transposed)
{
    __shared__ __align__(16) unsigned short As[64][40];   // 64 rows x 32 k (+pad)
    __shared__ __align__(16) unsigned short Bs[64][40];   // 64 cols x 32 k (W transposed)

    const int tid  = threadIdx.x;
    const int lane = tid & 31;
    const int wv   = tid >> 5;
    const int m0   = blockIdx.x * 64;
    const int n0   = blockIdx.y * 64;

    v8f acc[4];
    for (int ct = 0; ct < 4; ++ct)
        for (int i = 0; i < 8; ++i) acc[ct][i] = 0.0f;

    for (int k0 = 0; k0 < DMODEL; k0 += 32) {
        for (int it = 0; it < 4; ++it) {
            int i   = tid + it * 128;          // 512 float4s
            int row = i >> 3;
            int c4  = (i & 7) * 4;
            const float4 f = *(const float4*)&X[(size_t)(m0 + row) * DMODEL + k0 + c4];
            As[row][c4 + 0] = f2bf(f.x);
            As[row][c4 + 1] = f2bf(f.y);
            As[row][c4 + 2] = f2bf(f.z);
            As[row][c4 + 3] = f2bf(f.w);
        }
        for (int it = 0; it < 4; ++it) {
            int i  = tid + it * 128;           // 512 float4s (32 kk x 16 n-quads)
            int kk = i >> 4;
            int n4 = (i & 15) * 4;
            const float4 f = *(const float4*)&W[(size_t)(k0 + kk) * DMODEL + n0 + n4];
            Bs[n4 + 0][kk] = f2bf(f.x);
            Bs[n4 + 1][kk] = f2bf(f.y);
            Bs[n4 + 2][kk] = f2bf(f.z);
            Bs[n4 + 3][kk] = f2bf(f.w);
        }
        __syncthreads();

        BFrag a;
        {
            int arow = wv * 16 + (lane & 15);
            int akb  = (lane < 16) ? 0 : 8;
            a.u4[0] = *(const u32x4*)&As[arow][akb];
            a.u4[1] = *(const u32x4*)&As[arow][akb + 16];
        }
        for (int ct = 0; ct < 4; ++ct) {
            BFrag bf;
            int bcol = ct * 16 + (lane & 15);
            int bkb  = (lane < 16) ? 0 : 8;
            bf.u4[0] = *(const u32x4*)&Bs[bcol][bkb];
            bf.u4[1] = *(const u32x4*)&Bs[bcol][bkb + 16];
            acc[ct] = __builtin_amdgcn_wmma_f32_16x16x32_bf16(
                false, a.v, false, bf.v, (short)0, acc[ct], false, false);
        }
        __syncthreads();
    }

    const int hi8  = (lane < 16) ? 0 : 8;
    const int nlow = lane & 15;
    const int h    = blockIdx.y;
    for (int ct = 0; ct < 4; ++ct) {
        int d = ct * 16 + nlow;
        float bv = bias[n0 + d];
        for (int r = 0; r < 8; ++r) {
            int row = m0 + wv * 16 + r + hi8;   // = b*S + s
            int b   = row >> 12;
            int s   = row & (S_LEN - 1);
            float v = (acc[ct][r] + bv) * scale;
            size_t idx = transposed
                ? ((size_t)(b * NH + h) * DHEAD + d) * S_LEN + s
                : ((size_t)(b * NH + h) * S_LEN + s) * DHEAD + d;
            out[idx] = f2bf(v);
        }
    }
}

// ---------------------------------------------------------------------------
// Kernel 2: banded attention with online softmax. K/V windows bulk-loaded to
// LDS by the Tensor Data Mover; K, V and P WMMA operands come from LDS, Q from global.
// grid (16 chunks, 12 heads, 2 batch), block 256 (8 waves, 32 q-rows each)
// ---------------------------------------------------------------------------
__global__ __launch_bounds__(256)
void lf_attn_kernel(const unsigned short* __restrict__ qws,
                    const unsigned short* __restrict__ kws,
                    const unsigned short* __restrict__ vws,
                    const unsigned char*  __restrict__ mask,
                    float* __restrict__ out)
{
    __shared__ __align__(16) unsigned short Ks[NKEYS][DHEAD];   // 96 KB window of K
    __shared__ __align__(16) unsigned short Vs[DHEAD][NKEYS];   // 96 KB window of V^T
    __shared__ __align__(16) unsigned short Ps[8][32][72];      // 36 KB per-wave P bounce
    __shared__ float mAdd[NKEYS];

    const int chunk = blockIdx.x;
    const int h     = blockIdx.y;
    const int b     = blockIdx.z;
    const int tid   = threadIdx.x;
    const int lane  = tid & 31;
    const int wv    = tid >> 5;

    const int cbase = chunk * CHUNK;
    const int wbase = cbase - CHUNK;
    // window rows staged in LDS: always 768 fully-valid rows [row0, row0+768)
    const int row0  = min(max(wbase, 0), S_LEN - NKEYS);

    const unsigned short* qp = qws + (size_t)(b * NH + h) * S_LEN * DHEAD;
    const unsigned short* kp = kws + (size_t)(b * NH + h) * S_LEN * DHEAD;
    const unsigned short* vp = vws + (size_t)(b * NH + h) * DHEAD * S_LEN;

    // ---- TDM: one wave issues both window DMAs ----
    if (wv == 0) {
        // K window: 768 rows x 64 cols, contiguous rows (stride 64 elems)
        tdm_load_2d_b16(lds_byte_offset(&Ks[0][0]), kp + (size_t)row0 * DHEAD,
                        /*tile0=*/DHEAD, /*tile1=*/NKEYS,
                        /*stride0=*/DHEAD, /*dim0=*/DHEAD, /*dim1=*/NKEYS);
        // V^T window: 64 rows x 768 cols out of the [64][4096] transposed V
        tdm_load_2d_b16(lds_byte_offset(&Vs[0][0]), vp + row0,
                        /*tile0=*/NKEYS, /*tile1=*/DHEAD,
                        /*stride0=*/S_LEN, /*dim0=*/NKEYS, /*dim1=*/DHEAD);
        tdm_wait0();
    }

    for (int j = tid; j < NKEYS; j += 256) {
        int kabs = wbase + j;
        float v = 0.0f;
        if ((unsigned)kabs < (unsigned)S_LEN)
            v = mask[b * S_LEN + kabs] ? 0.0f : -10000.0f;
        mAdd[j] = v;
    }
    __syncthreads();

    const int nlow  = lane & 15;
    const int hi8   = (lane < 16) ? 0 : 8;
    const int qrow0 = wv * 32;

    // preload Q A-fragments [row-tile][k-step], K = DHEAD = 2 steps of 32
    BFrag aq[2][2];
    for (int rt = 0; rt < 2; ++rt) {
        int qabs = cbase + qrow0 + rt * 16 + nlow;
        const unsigned short* qr = qp + (size_t)qabs * DHEAD;
        for (int ks = 0; ks < 2; ++ks) {
            int kb = ks * 32 + ((lane < 16) ? 0 : 8);
            aq[rt][ks].u4[0] = *(const u32x4*)&qr[kb];
            aq[rt][ks].u4[1] = *(const u32x4*)&qr[kb + 16];
        }
    }

    float m_i[2][8], l_i[2][8];
    v8f o[2][4];
    for (int rt = 0; rt < 2; ++rt) {
        for (int r = 0; r < 8; ++r) { m_i[rt][r] = -1e30f; l_i[rt][r] = 0.0f; }
        for (int ds = 0; ds < 4; ++ds)
            for (int i = 0; i < 8; ++i) o[rt][ds][i] = 0.0f;
    }

    for (int kt = 0; kt < 12; ++kt) {
        // ---- scores S = Q K^T (32x64 per wave), K frags from LDS ----
        v8f sc[2][4];
        for (int rt = 0; rt < 2; ++rt)
            for (int ct = 0; ct < 4; ++ct)
                for (int i = 0; i < 8; ++i) sc[rt][ct][i] = 0.0f;

        for (int ks = 0; ks < 2; ++ks) {
            for (int ct = 0; ct < 4; ++ct) {
                int j    = kt * 64 + ct * 16 + nlow;
                int kabs = wbase + j;
                int krow = min(max(kabs - row0, 0), NKEYS - 1);  // masked later
                int d0   = ks * 32 + ((lane < 16) ? 0 : 8);
                BFrag bk;
                bk.u4[0] = *(const u32x4*)&Ks[krow][d0];
                bk.u4[1] = *(const u32x4*)&Ks[krow][d0 + 16];
                for (int rt = 0; rt < 2; ++rt)
                    sc[rt][ct] = __builtin_amdgcn_wmma_f32_16x16x32_bf16(
                        false, aq[rt][ks].v, false, bk.v, (short)0, sc[rt][ct], false, false);
            }
        }

        // ---- band/range mask + online softmax; P -> LDS (bf16) ----
        for (int rt = 0; rt < 2; ++rt) {
            for (int r = 0; r < 8; ++r) {
                int qabs = cbase + qrow0 + rt * 16 + r + hi8;
                float rm = -1e30f;
                float sv[4];
                for (int ct = 0; ct < 4; ++ct) {
                    int j    = kt * 64 + ct * 16 + nlow;
                    int kabs = wbase + j;
                    int dd   = kabs - qabs;
                    bool valid = ((unsigned)kabs < (unsigned)S_LEN) &&
                                 (dd >= -WINHALF) && (dd <= WINHALF);
                    float s = valid ? (sc[rt][ct][r] + mAdd[j]) : -1e30f;
                    sv[ct] = s;
                    rm = fmaxf(rm, s);
                }
                for (int off = 1; off < 16; off <<= 1)
                    rm = fmaxf(rm, __shfl_xor(rm, off, 16));
                float mnew = fmaxf(m_i[rt][r], rm);
                float corr = __expf(m_i[rt][r] - mnew);
                m_i[rt][r] = mnew;

                float rs = 0.0f;
                int prow = rt * 16 + r + hi8;
                for (int ct = 0; ct < 4; ++ct) {
                    float p = (sv[ct] > -1e29f) ? __expf(sv[ct] - mnew) : 0.0f;
                    rs += p;
                    Ps[wv][prow][ct * 16 + nlow] = f2bf(p);
                }
                for (int off = 1; off < 16; off <<= 1)
                    rs += __shfl_xor(rs, off, 16);
                l_i[rt][r] = l_i[rt][r] * corr + rs;
                for (int ds = 0; ds < 4; ++ds)
                    o[rt][ds][r] *= corr;
            }
        }

        // ---- O += P @ V, V frags from LDS ----
        for (int ks = 0; ks < 2; ++ks) {
            BFrag ap[2];
            for (int rt = 0; rt < 2; ++rt) {
                int prow = rt * 16 + nlow;
                int kb   = ks * 32 + ((lane < 16) ? 0 : 16);
                ap[rt].u4[0] = *(const u32x4*)&Ps[wv][prow][kb];
                ap[rt].u4[1] = *(const u32x4*)&Ps[wv][prow][kb + 8];
            }
            for (int ds = 0; ds < 4; ++ds) {
                int d     = ds * 16 + nlow;
                int j0    = kt * 64 + ks * 32 + ((lane < 16) ? 0 : 16);
                int j0l   = min(max(wbase + j0 - row0, 0), NKEYS - 16);
                BFrag bvv;
                bvv.u4[0] = *(const u32x4*)&Vs[d][j0l];
                bvv.u4[1] = *(const u32x4*)&Vs[d][j0l + 8];
                for (int rt = 0; rt < 2; ++rt)
                    o[rt][ds] = __builtin_amdgcn_wmma_f32_16x16x32_bf16(
                        false, ap[rt].v, false, bvv.v, (short)0, o[rt][ds], false, false);
            }
        }
    }

    // ---- normalize + store fp32 [b, s, h*64+d] ----
    for (int rt = 0; rt < 2; ++rt) {
        for (int r = 0; r < 8; ++r) {
            int qabs = cbase + qrow0 + rt * 16 + r + hi8;
            float l   = l_i[rt][r];
            float inv = (l > 0.0f) ? (1.0f / l) : 0.0f;
            float* orow = out + (size_t)(b * S_LEN + qabs) * DMODEL + h * DHEAD;
            for (int ds = 0; ds < 4; ++ds)
                orow[ds * 16 + nlow] = o[rt][ds][r] * inv;
        }
    }
}

// ---------------------------------------------------------------------------
extern "C" void kernel_launch(void* const* d_in, const int* in_sizes, int n_in,
                              void* d_out, int out_size, void* d_ws, size_t ws_size,
                              hipStream_t stream)
{
    const float*         query = (const float*)d_in[0];
    const unsigned char* mask  = (const unsigned char*)d_in[3];
    const float*         Wq    = (const float*)d_in[4];
    const float*         bq    = (const float*)d_in[5];
    const float*         Wk    = (const float*)d_in[6];
    const float*         bk    = (const float*)d_in[7];
    const float*         Wv    = (const float*)d_in[8];
    const float*         bv    = (const float*)d_in[9];

    unsigned short* qws = (unsigned short*)d_ws;
    const size_t QKV = (size_t)2 * NH * S_LEN * DHEAD;   // 6.29M bf16 each
    unsigned short* kws = qws + QKV;
    unsigned short* vws = kws + QKV;

    dim3 g1(128, 12), b1(128);
    qkv_proj_kernel<<<g1, b1, 0, stream>>>(query, Wq, bq, qws, 0.125f, 0); // 1/sqrt(64)
    qkv_proj_kernel<<<g1, b1, 0, stream>>>(query, Wk, bk, kws, 1.0f, 0);
    qkv_proj_kernel<<<g1, b1, 0, stream>>>(query, Wv, bv, vws, 1.0f, 1);   // V transposed

    dim3 g2(16, 12, 2), b2(256);
    lf_attn_kernel<<<g2, b2, 0, stream>>>(qws, kws, vws, mask, (float*)d_out);
}
